// RT1Transformer_34952443854875
// MI455X (gfx1250) — matrix operand order repaired and checked by reading, not compiled, not run on hardware
//
#include <hip/hip_runtime.h>
#include <stdint.h>

// ---- problem constants ----
#define DD 512
#define HH 8
#define LL 4
#define VV 256
#define BB 32
#define KK 32
#define VIZ 486
#define AA 66
#define SS (KK + VIZ + AA)   // 584
#define PPX (KK + VIZ)       // 518 (prefix length)
#define DH 64                // head dim

typedef unsigned short u16;
typedef __attribute__((ext_vector_type(16))) __bf16 v16bf;
typedef __attribute__((ext_vector_type(8)))  float  v8f;
typedef int v4i32 __attribute__((vector_size(16)));

union FragAB { v16bf v; uint4 q[2]; };
union FragC  { v8f v; float f[8]; };

__device__ __forceinline__ u16 f2bf(float f) {
    uint32_t u = __float_as_uint(f);
    return (u16)((u + 0x7fffu + ((u >> 16) & 1u)) >> 16);
}

// ---- gfx1250 async global->LDS staging (guarded; sync fallback) ----
#if defined(__has_builtin)
#if __has_builtin(__builtin_amdgcn_global_load_async_to_lds_b128)
#define HAVE_ASYNC_LDS 1
#endif
#endif
#ifndef HAVE_ASYNC_LDS
#define HAVE_ASYNC_LDS 0
#endif

#if HAVE_ASYNC_LDS
__device__ __forceinline__ void cp16_async(const u16* g, u16* l) {
    __builtin_amdgcn_global_load_async_to_lds_b128(
        (__attribute__((address_space(1))) v4i32*)(uintptr_t)g,
        (__attribute__((address_space(3))) v4i32*)l, 0, 0);
}
#if __has_builtin(__builtin_amdgcn_s_wait_asynccnt)
#define WAIT_ASYNC() __builtin_amdgcn_s_wait_asynccnt(0)
#else
#define WAIT_ASYNC() asm volatile("s_wait_asynccnt 0x0" ::: "memory")
#endif
#else
#define WAIT_ASYNC()
#endif

__device__ __forceinline__ void cp16(const u16* g, u16* l) {
#if HAVE_ASYNC_LDS
    cp16_async(g, l);
#else
    *(uint4*)l = *(const uint4*)g;
#endif
}

// =====================================================================
// All-bf16 WMMA GEMM:  C(M,N) = A(M,K) * W(N,K)^T + bias
//   A, W: bf16 row-major.  C: fp32 or bf16.  M%64==0, N%128==0, K%64==0.
// Block: 256 thr = 8 waves (2x4), tile 64x128, wave tile 32x32.
// K-step 64, double-buffered LDS, async global->LDS staging when available.
// =====================================================================
template <bool RELU, bool OUTBF16>
__global__ __launch_bounds__(256) void gemm_bf16_wmma(
    const u16* __restrict__ Ap, const u16* __restrict__ Wp,
    const float* __restrict__ bias, void* __restrict__ Cp,
    int M, int N, int Kd)
{
    __shared__ __align__(16) u16 As[2][64][72];   // 64 cols + 8 pad (144B rows)
    __shared__ __align__(16) u16 Bs[2][128][72];

    const int tid  = threadIdx.x;
    const int lane = tid & 31;
    const int wave = tid >> 5;
    const int wm   = wave >> 2;   // 0..1
    const int wn   = wave & 3;    // 0..3
    const int lh   = lane >> 4;
    const int l16  = lane & 15;
    const int blockM = blockIdx.y * 64;
    const int blockN = blockIdx.x * 128;

    FragC acc[2][2];
#pragma unroll
    for (int i = 0; i < 2; i++)
#pragma unroll
        for (int j = 0; j < 2; j++)
#pragma unroll
            for (int r = 0; r < 8; r++) acc[i][j].f[r] = 0.f;

    // staging: 16B segments; A tile 64x64 = 512 segs, B tile 128x64 = 1024 segs
    auto stage = [&](int bu, int k0) {
#pragma unroll
        for (int i = 0; i < 2; i++) {
            int idx = tid + i * 256;
            int row = idx >> 3, seg = (idx & 7) * 8;
            cp16(Ap + (int64_t)(blockM + row) * Kd + k0 + seg, &As[bu][row][seg]);
        }
#pragma unroll
        for (int i = 0; i < 4; i++) {
            int idx = tid + i * 256;
            int row = idx >> 3, seg = (idx & 7) * 8;
            cp16(Wp + (int64_t)(blockN + row) * Kd + k0 + seg, &Bs[bu][row][seg]);
        }
    };

    stage(0, 0);
    WAIT_ASYNC();
    __syncthreads();

    const int nk = Kd >> 6;
    int bu = 0;
    for (int kt = 0; kt < nk; kt++) {
        if (kt + 1 < nk) stage(bu ^ 1, (kt + 1) * 64);  // prefetch next tile
#pragma unroll
        for (int kk = 0; kk < 2; kk++) {                // two 32-wide k-steps
            FragAB af[2], bfr[2];
#pragma unroll
            for (int mt = 0; mt < 2; mt++) {
                int r = wm * 32 + mt * 16 + l16;
                af[mt].q[0] = *(const uint4*)&As[bu][r][kk * 32 + lh * 8];
                af[mt].q[1] = *(const uint4*)&As[bu][r][kk * 32 + 16 + lh * 8];
            }
#pragma unroll
            for (int nt = 0; nt < 2; nt++) {
                int r = wn * 32 + nt * 16 + l16;
                bfr[nt].q[0] = *(const uint4*)&Bs[bu][r][kk * 32 + lh * 8];
                bfr[nt].q[1] = *(const uint4*)&Bs[bu][r][kk * 32 + 16 + lh * 8];
            }
#pragma unroll
            for (int mt = 0; mt < 2; mt++)
#pragma unroll
                for (int nt = 0; nt < 2; nt++)
                    acc[mt][nt].v = __builtin_amdgcn_wmma_f32_16x16x32_bf16(
                        false, af[mt].v, false, bfr[nt].v, (short)0,
                        acc[mt][nt].v, false, false);
        }
        WAIT_ASYNC();
        __syncthreads();
        bu ^= 1;
    }

    // ---- epilogue ----
#pragma unroll
    for (int mt = 0; mt < 2; mt++) {
#pragma unroll
        for (int nt = 0; nt < 2; nt++) {
            int gc = blockN + wn * 32 + nt * 16 + l16;
            float bv = bias[gc];
#pragma unroll
            for (int r = 0; r < 8; r++) {
                int gr = blockM + wm * 32 + mt * 16 + lh * 8 + r;
                float v = acc[mt][nt].f[r] + bv;
                if (RELU) v = fmaxf(v, 0.f);
                int64_t o = (int64_t)gr * N + gc;
                if (OUTBF16) ((u16*)Cp)[o] = f2bf(v);
                else ((float*)Cp)[o] = v;
            }
        }
    }
}

// =====================================================================
// Flash-style attention, 1 wave per (b, h, 16-query tile).
// qkv: (B, S, 3D) bf16.  out: (B, S, D) bf16.
// Prefix-causal mask: allowed(i,j) = (j < P) || (j <= i).
// =====================================================================
__global__ __launch_bounds__(32) void attn_flash(
    const u16* __restrict__ qkv, u16* __restrict__ outp)
{
    __shared__ __align__(16) u16 Vt[64][32];  // transposed V tile [d][key]
    __shared__ __align__(16) u16 Pb[16][32];  // softmax P tile

    const int lane = threadIdx.x & 31;
    const int lh   = lane >> 4;
    const int l16  = lane & 15;
    const int qt   = blockIdx.x;
    const int bh   = blockIdx.y;
    const int b    = bh / HH, h = bh % HH;
    const int q0   = qt * 16;
    const int64_t bS = (int64_t)b * SS;
    const float scale = 0.125f;  // 1/sqrt(64)

    FragAB qf[2];
    {
        int sq = q0 + l16; if (sq > SS - 1) sq = SS - 1;
        const u16* qsrc = qkv + (bS + sq) * (3 * DD) + h * DH;
#pragma unroll
        for (int ks = 0; ks < 2; ks++) {
            qf[ks].q[0] = *(const uint4*)(qsrc + ks * 32 + lh * 8);
            qf[ks].q[1] = *(const uint4*)(qsrc + ks * 32 + 16 + lh * 8);
        }
    }

    FragC accO[4];
#pragma unroll
    for (int nf = 0; nf < 4; nf++)
#pragma unroll
        for (int r = 0; r < 8; r++) accO[nf].f[r] = 0.f;
    float rm[8], rs[8];
#pragma unroll
    for (int r = 0; r < 8; r++) { rm[r] = -3.0e38f; rs[r] = 0.f; }

    const int NT = (SS + 31) / 32;  // 19 key tiles of 32
    for (int jt = 0; jt < NT; jt++) {
        const int j0 = jt * 32;
        __syncthreads();
        {   // stage V tile transposed: Vt[d][key]
            int key = j0 + lane; if (key > SS - 1) key = SS - 1;
            const u16* vsrc = qkv + (bS + key) * (3 * DD) + 2 * DD + h * DH;
#pragma unroll
            for (int d8 = 0; d8 < 64; d8 += 8) {
                uint4 t = *(const uint4*)(vsrc + d8);
                Vt[d8 + 0][lane] = (u16)t.x; Vt[d8 + 1][lane] = (u16)(t.x >> 16);
                Vt[d8 + 2][lane] = (u16)t.y; Vt[d8 + 3][lane] = (u16)(t.y >> 16);
                Vt[d8 + 4][lane] = (u16)t.z; Vt[d8 + 5][lane] = (u16)(t.z >> 16);
                Vt[d8 + 6][lane] = (u16)t.w; Vt[d8 + 7][lane] = (u16)(t.w >> 16);
            }
        }
        FragC s1, s2;
#pragma unroll
        for (int r = 0; r < 8; r++) { s1.f[r] = 0.f; s2.f[r] = 0.f; }
#pragma unroll
        for (int nt = 0; nt < 2; nt++) {
            int key = j0 + nt * 16 + l16; if (key > SS - 1) key = SS - 1;
            const u16* ksrc = qkv + (bS + key) * (3 * DD) + DD + h * DH;
#pragma unroll
            for (int ks = 0; ks < 2; ks++) {
                FragAB kf;
                kf.q[0] = *(const uint4*)(ksrc + ks * 32 + lh * 8);
                kf.q[1] = *(const uint4*)(ksrc + ks * 32 + 16 + lh * 8);
                if (nt == 0)
                    s1.v = __builtin_amdgcn_wmma_f32_16x16x32_bf16(
                        false, qf[ks].v, false, kf.v, (short)0, s1.v, false, false);
                else
                    s2.v = __builtin_amdgcn_wmma_f32_16x16x32_bf16(
                        false, qf[ks].v, false, kf.v, (short)0, s2.v, false, false);
            }
        }
#pragma unroll
        for (int r = 0; r < 8; r++) {
            int i  = q0 + lh * 8 + r;
            int c1 = j0 + l16, c2 = c1 + 16;
            float v1 = s1.f[r] * scale + (((c1 < PPX) || (c1 <= i)) ? 0.f : -1e9f);
            float v2 = s2.f[r] * scale + (((c2 < PPX) || (c2 <= i)) ? 0.f : -1e9f);
            float mx = fmaxf(v1, v2);
#pragma unroll
            for (int off = 8; off > 0; off >>= 1) mx = fmaxf(mx, __shfl_xor(mx, off, 32));
            float nm = fmaxf(rm[r], mx);
            float al = __expf(rm[r] - nm);
            float p1 = __expf(v1 - nm);
            float p2 = __expf(v2 - nm);
            float ps = p1 + p2;
#pragma unroll
            for (int off = 8; off > 0; off >>= 1) ps += __shfl_xor(ps, off, 32);
            rs[r] = rs[r] * al + ps;
            rm[r] = nm;
#pragma unroll
            for (int nf = 0; nf < 4; nf++) accO[nf].f[r] *= al;
            Pb[lh * 8 + r][l16]      = f2bf(p1);
            Pb[lh * 8 + r][16 + l16] = f2bf(p2);
        }
        __syncthreads();
        FragAB pA;
        pA.q[0] = *(const uint4*)&Pb[l16][lh * 8];
        pA.q[1] = *(const uint4*)&Pb[l16][16 + lh * 8];
#pragma unroll
        for (int nf = 0; nf < 4; nf++) {
            FragAB vB;
            int dc = nf * 16 + l16;
            vB.q[0] = *(const uint4*)&Vt[dc][lh * 8];
            vB.q[1] = *(const uint4*)&Vt[dc][16 + lh * 8];
            accO[nf].v = __builtin_amdgcn_wmma_f32_16x16x32_bf16(
                false, pA.v, false, vB.v, (short)0, accO[nf].v, false, false);
        }
    }
#pragma unroll
    for (int r = 0; r < 8; r++) {
        int i = q0 + lh * 8 + r;
        if (i < SS) {
            float inv = 1.f / rs[r];
#pragma unroll
            for (int nf = 0; nf < 4; nf++)
                outp[(bS + i) * DD + h * DH + nf * 16 + l16] =
                    f2bf(accO[nf].f[r] * inv);
        }
    }
}

// ---- fp32 -> bf16 weight conversion (n % 8 == 0) ----
__global__ __launch_bounds__(256) void cvt_f32_bf16(
    const float* __restrict__ src, u16* __restrict__ dst, int n)
{
    int i = (blockIdx.x * 256 + threadIdx.x) * 8;
    if (i < n) {
        float4 a = *(const float4*)(src + i);
        float4 b = *(const float4*)(src + i + 4);
        uint4 t;
        t.x = (uint32_t)f2bf(a.x) | ((uint32_t)f2bf(a.y) << 16);
        t.y = (uint32_t)f2bf(a.z) | ((uint32_t)f2bf(a.w) << 16);
        t.z = (uint32_t)f2bf(b.x) | ((uint32_t)f2bf(b.y) << 16);
        t.w = (uint32_t)f2bf(b.z) | ((uint32_t)f2bf(b.w) << 16);
        *(uint4*)(dst + i) = t;
    }
}

// ---- embedding/concat: x[b,s,:] (fp32 + bf16 copies) ----
__global__ __launch_bounds__(256) void embed_kernel(
    const float* __restrict__ lang, const float* __restrict__ vis,
    const int* __restrict__ toks, const float* __restrict__ emb,
    float* __restrict__ x, u16* __restrict__ xb)
{
    const int t = blockIdx.x;
    const int b = t / SS, s = t % SS;
    const float* src;
    if (s < KK)       src = lang + (int64_t)(b * KK + s) * DD;
    else if (s < PPX) src = vis + (int64_t)(b * VIZ + (s - KK)) * DD;
    else              src = emb + (int64_t)toks[b * AA + (s - PPX)] * DD;
    float* dst = x + (int64_t)t * DD;
    u16*  dstb = xb + (int64_t)t * DD;
    const int c = threadIdx.x;
    float v0 = src[c], v1 = src[c + 256];
    dst[c] = v0; dst[c + 256] = v1;
    dstb[c] = f2bf(v0); dstb[c + 256] = f2bf(v1);
}

// ---- fused residual add + LayerNorm (fp32 stream + bf16 copy) ----
__global__ __launch_bounds__(256) void add_ln_kernel(
    float* __restrict__ x, u16* __restrict__ xb, const float* __restrict__ o,
    const float* __restrict__ g, const float* __restrict__ be)
{
    __shared__ float red[256];
    const int t = blockIdx.x;
    const int c = threadIdx.x;
    float* row = x + (int64_t)t * DD;
    u16*  rowb = xb + (int64_t)t * DD;
    const float* orow = o + (int64_t)t * DD;
    float v0 = row[c] + orow[c];
    float v1 = row[c + 256] + orow[c + 256];
    red[c] = v0 + v1;
    for (int s2 = 128; s2 > 0; s2 >>= 1) { __syncthreads(); if (c < s2) red[c] += red[c + s2]; }
    __syncthreads();
    float mu = red[0] * (1.f / 512.f);
    __syncthreads();
    float d0 = v0 - mu, d1 = v1 - mu;
    red[c] = d0 * d0 + d1 * d1;
    for (int s2 = 128; s2 > 0; s2 >>= 1) { __syncthreads(); if (c < s2) red[c] += red[c + s2]; }
    __syncthreads();
    float inv = rsqrtf(red[0] * (1.f / 512.f) + 1e-5f);
    float y0 = d0 * inv * g[c] + be[c];
    float y1 = d1 * inv * g[c + 256] + be[c + 256];
    row[c] = y0; row[c + 256] = y1;
    rowb[c] = f2bf(y0); rowb[c + 256] = f2bf(y1);
}

// ---- gather action-token rows (bf16 -> contiguous bf16) ----
__global__ __launch_bounds__(256) void gather_act(
    const u16* __restrict__ xb, u16* __restrict__ out)
{
    const int t = blockIdx.x;  // b*AA + a
    const int b = t / AA, a = t % AA;
    const u16* src = xb + (int64_t)(b * SS + PPX + a) * DD;
    u16* dst = out + (int64_t)t * DD;
    const int c = threadIdx.x;
    dst[c] = src[c];
    dst[c + 256] = src[c + 256];
}

extern "C" void kernel_launch(void* const* d_in, const int* in_sizes, int n_in,
                              void* d_out, int out_size, void* d_ws, size_t ws_size,
                              hipStream_t stream)
{
    (void)in_sizes; (void)n_in; (void)out_size; (void)ws_size;
    const float* lang = (const float*)d_in[0];
    const float* vis  = (const float*)d_in[1];
    const int*   toks = (const int*)d_in[2];
    const float* emb  = (const float*)d_in[3];
    const float* Wqkv = (const float*)d_in[4];
    const float* bqkv = (const float*)d_in[5];
    const float* Wo   = (const float*)d_in[6];
    const float* bo   = (const float*)d_in[7];
    const float* W1   = (const float*)d_in[8];
    const float* b1   = (const float*)d_in[9];
    const float* W2   = (const float*)d_in[10];
    const float* b2   = (const float*)d_in[11];
    const float* ln1g = (const float*)d_in[12];
    const float* ln1b = (const float*)d_in[13];
    const float* ln2g = (const float*)d_in[14];
    const float* ln2b = (const float*)d_in[15];
    const float* hW   = (const float*)d_in[16];
    const float* hb   = (const float*)d_in[17];

    const int BS = BB * SS;  // 18688 (divisible by 64)
    char* ws = (char*)d_ws;
    size_t off = 0;
    auto alloc = [&](size_t bytes) -> void* {
        void* p = ws + off; off += (bytes + 255) & ~(size_t)255; return p;
    };
    float* x      = (float*)alloc((size_t)BS * DD * 4);
    u16*   xb     = (u16*)  alloc((size_t)BS * DD * 2);
    u16*   big    = (u16*)  alloc((size_t)BS * 4 * DD * 2);   // qkv / ffn hidden
    u16*   attnb  = (u16*)  alloc((size_t)BS * DD * 2);
    float* of     = (float*)alloc((size_t)BS * DD * 4);
    u16*   actb   = (u16*)  alloc((size_t)BB * AA * DD * 2);
    u16*   wqkv_b = (u16*)  alloc((size_t)LL * 3 * DD * DD * 2);
    u16*   wo_b   = (u16*)  alloc((size_t)LL * DD * DD * 2);
    u16*   w1_b   = (u16*)  alloc((size_t)LL * 4 * DD * DD * 2);
    u16*   w2_b   = (u16*)  alloc((size_t)LL * 4 * DD * DD * 2);
    u16*   hw_b   = (u16*)  alloc((size_t)VV * DD * 2);

    // one-time (per launch) weight conversion fp32 -> bf16
    {
        int n;
        n = LL * 3 * DD * DD; cvt_f32_bf16<<<n / 2048, 256, 0, stream>>>(Wqkv, wqkv_b, n);
        n = LL * DD * DD;     cvt_f32_bf16<<<n / 2048, 256, 0, stream>>>(Wo, wo_b, n);
        n = LL * 4 * DD * DD; cvt_f32_bf16<<<n / 2048, 256, 0, stream>>>(W1, w1_b, n);
        n = LL * 4 * DD * DD; cvt_f32_bf16<<<n / 2048, 256, 0, stream>>>(W2, w2_b, n);
        n = VV * DD;          cvt_f32_bf16<<<n / 2048, 256, 0, stream>>>(hW, hw_b, n);
    }

    embed_kernel<<<BS, 256, 0, stream>>>(lang, vis, toks, emb, x, xb);

    for (int l = 0; l < LL; l++) {
        const u16* Wql = wqkv_b + (size_t)l * 3 * DD * DD;
        const u16* Wol = wo_b + (size_t)l * DD * DD;
        const u16* W1l = w1_b + (size_t)l * 4 * DD * DD;
        const u16* W2l = w2_b + (size_t)l * DD * 4 * DD;
        const float* bql = bqkv + (size_t)l * 3 * DD;
        const float* bol = bo + (size_t)l * DD;
        const float* b1l = b1 + (size_t)l * 4 * DD;
        const float* b2l = b2 + (size_t)l * DD;

        gemm_bf16_wmma<false, true><<<dim3((3 * DD) / 128, BS / 64), 256, 0, stream>>>(
            xb, Wql, bql, big, BS, 3 * DD, DD);
        attn_flash<<<dim3((SS + 15) / 16, BB * HH), 32, 0, stream>>>(big, attnb);
        gemm_bf16_wmma<false, false><<<dim3(DD / 128, BS / 64), 256, 0, stream>>>(
            attnb, Wol, bol, of, BS, DD, DD);
        add_ln_kernel<<<BS, 256, 0, stream>>>(x, xb, of, ln1g + l * DD, ln1b + l * DD);
        gemm_bf16_wmma<true, true><<<dim3((4 * DD) / 128, BS / 64), 256, 0, stream>>>(
            xb, W1l, b1l, big, BS, 4 * DD, DD);
        gemm_bf16_wmma<false, false><<<dim3(DD / 128, BS / 64), 256, 0, stream>>>(
            big, W2l, b2l, of, BS, DD, 4 * DD);
        add_ln_kernel<<<BS, 256, 0, stream>>>(x, xb, of, ln2g + l * DD, ln2b + l * DD);
    }

    gather_act<<<BB * AA, 256, 0, stream>>>(xb, actb);
    gemm_bf16_wmma<false, false><<<dim3(VV / 128, (BB * AA) / 64), 256, 0, stream>>>(
        actb, hw_b, hb, (float*)d_out, BB * AA, VV, DD);
}
